// RGATLayer_49392123904431
// MI455X (gfx1250) — matrix operand order repaired and checked by reading, not compiled, not run on hardware
//
#include <hip/hip_runtime.h>
#include <stdint.h>
#include <stddef.h>

#define BSZ    4
#define SEQ    512
#define HID    512
#define NHEAD  8
#define DHEAD  64
#define NTOK   (BSZ * SEQ)        /* 2048 */
#define INV_SCALE 0.125f          /* 1/sqrt(64) */
#define LN_EPS 1e-5f
#define ROWSTRIDE 68              /* 64 f32 + 4 dword TDM pad -> spreads LDS banks */

typedef __attribute__((ext_vector_type(16))) __bf16 v16bf;
typedef __attribute__((ext_vector_type(8)))  float  v8f;
typedef unsigned int u32x4 __attribute__((ext_vector_type(4)));
typedef int          i32x8 __attribute__((ext_vector_type(8)));
typedef int          i32x4 __attribute__((ext_vector_type(4)));

union Frag16 { v16bf v; uint16_t u[16]; };

__device__ __forceinline__ uint16_t f2bf(float f) {
  union { float f; uint32_t u; } c; c.f = f;
  return (uint16_t)((c.u + 0x7FFFu + ((c.u >> 16) & 1u)) >> 16); // RNE
}

#if __has_builtin(__builtin_amdgcn_tensor_load_to_lds)
#define HAVE_TDM 1
#else
#define HAVE_TDM 0
#endif

#if HAVE_TDM
// TDM 2D tile load: `rows` x 64 f32 from gsrc (row stride 64 f32) into LDS at
// lds_byte_off, padding +4 dwords per 64-dword row (D# pad feature) so LDS
// rows land on ROWSTRIDE=68 dword pitch.
__device__ __forceinline__ void tdm_load_rows_f32(uint32_t lds_byte_off,
                                                  const float* gsrc, int rows) {
  union { u32x4 v; uint32_t a[4]; } g0 = {};
  union { i32x8 v; uint32_t a[8]; } g1 = {};
  const uint64_t ga = (uint64_t)(uintptr_t)gsrc;
  g0.a[0] = 1u;                                              // count=1 (valid D#)
  g0.a[1] = lds_byte_off;                                    // lds_addr
  g0.a[2] = (uint32_t)ga;                                    // global_addr[31:0]
  g0.a[3] = (uint32_t)((ga >> 32) & 0x1FFFFFFu) | (2u << 30);// addr[56:32] | type=2
  // data_size=4B(code2), pad_enable, pad_interval=64dw(code5), pad_amount=4dw(code3)
  g1.a[0] = (2u << 16) | (1u << 20) | (5u << 22) | (3u << 25);
  g1.a[1] = ((uint32_t)DHEAD & 0xFFFFu) << 16;               // tensor_dim0 lo16
  g1.a[2] = ((uint32_t)SEQ & 0xFFFFu) << 16;                 // dim0 hi16=0 | tensor_dim1 lo16
  g1.a[3] = ((uint32_t)DHEAD) << 16;                         // dim1 hi16=0 | tile_dim0=64
  g1.a[4] = (uint32_t)rows;                                  // tile_dim1 | tile_dim2=0
  g1.a[5] = (uint32_t)DHEAD;                                 // tensor_dim0_stride lo32
  g1.a[6] = 0u;                                              // stride0 hi | stride1 lo
  g1.a[7] = 0u;                                              // stride1 hi
  i32x4 z4 = {};
#if defined(__clang_major__) && (__clang_major__ >= 23)
  i32x8 z8 = {};
  __builtin_amdgcn_tensor_load_to_lds(g0.v, g1.v, z4, z4, z8, 0);
#else
  __builtin_amdgcn_tensor_load_to_lds(g0.v, g1.v, z4, z4, 0);
#endif
}
#endif

// ---------------- vectorized f32 -> bf16 (4 elems/thread) ----------------
__global__ void k_cvt_bf16_v4(const float4* __restrict__ src, uint2* __restrict__ dst, int n4) {
  int i = blockIdx.x * blockDim.x + threadIdx.x;
  if (i < n4) {
    const float4 f = src[i];
    uint2 o;
    o.x = (uint32_t)f2bf(f.x) | ((uint32_t)f2bf(f.y) << 16);
    o.y = (uint32_t)f2bf(f.z) | ((uint32_t)f2bf(f.w) << 16);
    dst[i] = o;
  }
}

// ---------------- f32 (K x N) -> bf16 transposed (N x K) ----------------
__global__ void k_cvt_bf16_t(const float* __restrict__ src, uint16_t* __restrict__ dst,
                             int K, int N) {
  long long idx = (long long)blockIdx.x * blockDim.x + threadIdx.x; // dst index
  int k = (int)(idx % K);
  long long n = idx / K;
  dst[idx] = f2bf(src[(long long)k * N + n]);
}

// ---------------- generic batched bf16 WMMA GEMM ----------------
// C[M,N] = A[M,K](bf16 row-major) * B(bf16, stored N x K) + bias, opt relu.
// One wave computes a 32x64 tile (2 A-frags x 4 B-frags = 8 WMMA / K-step),
// rotated software pipeline: next-step fragments are loaded before the current
// WMMAs issue, last step peeled so no dead loads / conditional in the loop.
struct GemmP {
  const uint16_t* A; const uint16_t* B;
  float* C; uint16_t* Cb;
  const float* bias;
  long long sA, sB, sC, sCb;   // batch strides in elements
  int lda, ldb, ldc, ldcb;
  int K, relu;
};

__device__ __forceinline__ void load_a(Frag16& f, const uint16_t* ap, int k0) {
#pragma unroll
  for (int t = 0; t < 8; ++t) { f.u[t] = ap[k0 + t]; f.u[8 + t] = ap[k0 + 8 + t]; }
}
__device__ __forceinline__ void load_b16(Frag16& f, const uint16_t* bp) {
#pragma unroll
  for (int t = 0; t < 16; ++t) f.u[t] = bp[t];
}
#define WMMA_BF16(A_, B_, C_) \
  __builtin_amdgcn_wmma_f32_16x16x32_bf16(false, (A_).v, false, (B_).v, (short)0, (C_), false, false)

__global__ void __launch_bounds__(32, 1) k_gemm_bf16(GemmP p) {
  const int lane = threadIdx.x & 31;
  const int hi = lane >> 4, ln = lane & 15;
  const int n0 = blockIdx.x << 6;   // 64 columns per wave
  const int m0 = blockIdx.y << 5;   // 32 rows per wave
  const uint16_t* A = p.A + (long long)blockIdx.z * p.sA;
  const uint16_t* B = p.B + (long long)blockIdx.z * p.sB;
  v8f acc[2][4] = {};
  const uint16_t* ap0 = A + (long long)(m0 + ln) * p.lda + hi * 16;
  const uint16_t* ap1 = ap0 + (long long)16 * p.lda;
  const uint16_t* bp0 = B + (long long)(n0 + ln) * p.ldb + hi * 16;
  const uint16_t* bp1 = bp0 + (long long)16 * p.ldb;
  const uint16_t* bp2 = bp0 + (long long)32 * p.ldb;
  const uint16_t* bp3 = bp0 + (long long)48 * p.ldb;

  Frag16 a0, a1, b0, b1, b2, b3;
  load_a(a0, ap0, 0); load_a(a1, ap1, 0);
  load_b16(b0, bp0); load_b16(b1, bp1); load_b16(b2, bp2); load_b16(b3, bp3);
  for (int k0 = 0; k0 + 32 < p.K; k0 += 32) {
    const int kn = k0 + 32;
    Frag16 na0, na1, nb0, nb1, nb2, nb3;
    load_a(na0, ap0, kn); load_a(na1, ap1, kn);
    load_b16(nb0, bp0 + kn); load_b16(nb1, bp1 + kn);
    load_b16(nb2, bp2 + kn); load_b16(nb3, bp3 + kn);
    acc[0][0] = WMMA_BF16(a0, b0, acc[0][0]);
    acc[1][0] = WMMA_BF16(a1, b0, acc[1][0]);
    acc[0][1] = WMMA_BF16(a0, b1, acc[0][1]);
    acc[1][1] = WMMA_BF16(a1, b1, acc[1][1]);
    acc[0][2] = WMMA_BF16(a0, b2, acc[0][2]);
    acc[1][2] = WMMA_BF16(a1, b2, acc[1][2]);
    acc[0][3] = WMMA_BF16(a0, b3, acc[0][3]);
    acc[1][3] = WMMA_BF16(a1, b3, acc[1][3]);
    a0 = na0; a1 = na1; b0 = nb0; b1 = nb1; b2 = nb2; b3 = nb3;
  }
  acc[0][0] = WMMA_BF16(a0, b0, acc[0][0]);
  acc[1][0] = WMMA_BF16(a1, b0, acc[1][0]);
  acc[0][1] = WMMA_BF16(a0, b1, acc[0][1]);
  acc[1][1] = WMMA_BF16(a1, b1, acc[1][1]);
  acc[0][2] = WMMA_BF16(a0, b2, acc[0][2]);
  acc[1][2] = WMMA_BF16(a1, b2, acc[1][2]);
  acc[0][3] = WMMA_BF16(a0, b3, acc[0][3]);
  acc[1][3] = WMMA_BF16(a1, b3, acc[1][3]);

  const bool hasC  = p.C  != nullptr;
  const bool hasCb = p.Cb != nullptr;
  float* Cp  = hasC  ? p.C  + (long long)blockIdx.z * p.sC  : nullptr;
  uint16_t* Cbp = hasCb ? p.Cb + (long long)blockIdx.z * p.sCb : nullptr;
#pragma unroll
  for (int mt = 0; mt < 2; ++mt) {
#pragma unroll
    for (int nt = 0; nt < 4; ++nt) {
      const int cn = n0 + nt * 16 + ln;
      const float bsv = p.bias ? p.bias[cn] : 0.0f;
#pragma unroll
      for (int r = 0; r < 8; ++r) {
        const int cm = m0 + mt * 16 + hi * 8 + r;
        float v = acc[mt][nt][r] + bsv;
        if (p.relu) v = fmaxf(v, 0.0f);
        if (hasC)  Cp [(long long)cm * p.ldc  + cn] = v;
        if (hasCb) Cbp[(long long)cm * p.ldcb + cn] = f2bf(v);
      }
    }
  }
}

// ---------------- split QKV: Q,K -> (b,h,i,d); V -> transposed (b,h,d,i) ----------------
__global__ void k_repack_qkv(const uint16_t* __restrict__ qkv, uint16_t* __restrict__ Q,
                             uint16_t* __restrict__ K, uint16_t* __restrict__ Vt) {
  long long idx = (long long)blockIdx.x * blockDim.x + threadIdx.x; // NTOK*3*HID
  int c = (int)(idx % (3 * HID));
  long long t = idx / (3 * HID);
  int b = (int)(t / SEQ), i = (int)(t % SEQ);
  int which = c / HID, cc = c % HID;
  int h = cc / DHEAD, d = cc % DHEAD;
  uint16_t val = qkv[idx];
  if (which == 2) {
    Vt[(((long long)b * NHEAD + h) * DHEAD + d) * SEQ + i] = val;
  } else {
    long long o = (((long long)b * NHEAD + h) * SEQ + i) * DHEAD + d;
    if (which == 0) Q[o] = val; else K[o] = val;
  }
}

// ---------------- scores += Q(8x64) . rel_k[b,i]^T : wave = (b,i, 4 j-tiles) ----------------
__global__ void k_rel_scores(const uint16_t* __restrict__ Q, const float* __restrict__ relk,
                             float* __restrict__ scores) {
  const int lane = threadIdx.x & 31;
  const int hi = lane >> 4, ln = lane & 15;
  const int j0 = blockIdx.x << 6;
  const int i  = blockIdx.y;
  const int b  = blockIdx.z;
  const float* rk = relk + (((long long)b * SEQ + i) * SEQ + (j0 + ln)) * DHEAD;
  v8f acc[4] = {};
#pragma unroll
  for (int k0 = 0; k0 < DHEAD; k0 += 32) {
    Frag16 a;
    if (ln < NHEAD) {   // A rows = heads (8 used, 8 zero-padded)
      const uint16_t* qp = Q + (((long long)b * NHEAD + ln) * SEQ + i) * DHEAD + k0 + hi * 16;
#pragma unroll
      for (int t = 0; t < 8; ++t) { a.u[t] = qp[t]; a.u[8 + t] = qp[8 + t]; }
    } else {
#pragma unroll
      for (int t = 0; t < 16; ++t) a.u[t] = 0;
    }
#pragma unroll
    for (int jt = 0; jt < 4; ++jt) {
      Frag16 bf;
      const float* rp = rk + (long long)(jt * 16) * DHEAD + k0 + hi * 16;
      __builtin_prefetch(rp + 4 * DHEAD, 0, 0);     // global_prefetch_b8 on the stream
#pragma unroll
      for (int t = 0; t < 16; ++t) bf.u[t] = f2bf(rp[t]);
      acc[jt] = __builtin_amdgcn_wmma_f32_16x16x32_bf16(false, a.v, false, bf.v,
                                                        (short)0, acc[jt], false, false);
    }
  }
  if (hi == 0) {
#pragma unroll
    for (int jt = 0; jt < 4; ++jt) {
      const int j = j0 + jt * 16 + ln;
#pragma unroll
      for (int r = 0; r < NHEAD; ++r)
        scores[(((long long)b * NHEAD + r) * SEQ + i) * SEQ + j] += acc[jt][r];
    }
  }
}

// ---------------- masked softmax over j, emit bf16 attention ----------------
__global__ void k_softmax(const float* __restrict__ scores, const unsigned char* __restrict__ mask,
                          uint16_t* __restrict__ Abf) {
  const int lane = threadIdx.x & 31;
  const int wv = threadIdx.x >> 5;
  const long long row = (long long)blockIdx.x * (blockDim.x >> 5) + wv; // (b*NH+h)*SEQ + i
  const int i = (int)(row % SEQ);
  const int b = (int)(row / ((long long)SEQ * NHEAD));
  const float* e = scores + row * SEQ;
  const unsigned char* mr = mask + ((long long)b * SEQ + i) * SEQ;
  float v[16];
  float mx = -3.0e38f;
#pragma unroll
  for (int t = 0; t < 16; ++t) {
    const int j = lane + t * 32;
    float x = e[j] * INV_SCALE;
    if (mr[j]) x = -1.0e20f;
    v[t] = x; mx = fmaxf(mx, x);
  }
#pragma unroll
  for (int off = 16; off > 0; off >>= 1) mx = fmaxf(mx, __shfl_xor(mx, off, 32));
  float s = 0.0f;
#pragma unroll
  for (int t = 0; t < 16; ++t) { v[t] = __expf(v[t] - mx); s += v[t]; }
#pragma unroll
  for (int off = 16; off > 0; off >>= 1) s += __shfl_xor(s, off, 32);
  const float inv = 1.0f / s;
  uint16_t* ar = Abf + row * SEQ;
#pragma unroll
  for (int t = 0; t < 16; ++t) ar[lane + t * 32] = f2bf(v[t] * inv);
}

// ---------------- out += A(8x512) . rel_v[b,i] ----------------
// One 128-thread workgroup per (b,i); wave w owns d-tile w. rel_v[b,i] is
// streamed by the Tensor Data Mover into double-buffered LDS (32x64 f32 tiles,
// TDM-padded to 68-dword rows), synchronized with s_wait_tensorcnt + barrier.
__global__ void __launch_bounds__(128) k_rel_out(const uint16_t* __restrict__ Abf,
                                                 const float* __restrict__ relv,
                                                 float* __restrict__ attnout) {
  __shared__ float lds[2][32 * ROWSTRIDE];
  const int i = blockIdx.x;
  const int b = blockIdx.y;
  const int lane = threadIdx.x & 31;
  const int hi = lane >> 4, ln = lane & 15;
  const int d0 = (threadIdx.x >> 5) << 4;   // wave's d-tile
  const float* rv = relv + ((long long)b * SEQ + i) * (long long)SEQ * DHEAD;
  v8f acc = {};
  const int NCH = SEQ / 32;                 // 16 chunks of 32 rows
#if HAVE_TDM
  if (threadIdx.x < 32)
    tdm_load_rows_f32((uint32_t)(uintptr_t)&lds[0][0], rv, 32);
#endif
  for (int t = 0; t < NCH; ++t) {
    const float* buf = lds[t & 1];
#if HAVE_TDM
    if (threadIdx.x < 32) {
      if (t + 1 < NCH) {
        tdm_load_rows_f32((uint32_t)(uintptr_t)&lds[(t + 1) & 1][0],
                          rv + (long long)(t + 1) * 32 * DHEAD, 32);
        __builtin_amdgcn_s_wait_tensorcnt(1);   // chunk t complete, t+1 in flight
      } else {
        __builtin_amdgcn_s_wait_tensorcnt(0);
      }
    }
    __syncthreads();                            // publish chunk t to all waves
#else
    float* wbuf = lds[t & 1];
    for (int e = threadIdx.x; e < 32 * DHEAD; e += 128) {
      const int r = e >> 6, c = e & 63;
      wbuf[r * ROWSTRIDE + c] = rv[(long long)t * 32 * DHEAD + r * DHEAD + c];
    }
    __syncthreads();
#endif
    Frag16 a, bf;
    const int k0 = t * 32;
    if (ln < NHEAD) {
      const uint16_t* apx = Abf + (((long long)b * NHEAD + ln) * SEQ + i) * SEQ + k0 + hi * 16;
#pragma unroll
      for (int tt = 0; tt < 8; ++tt) { a.u[tt] = apx[tt]; a.u[8 + tt] = apx[8 + tt]; }
    } else {
#pragma unroll
      for (int tt = 0; tt < 16; ++tt) a.u[tt] = 0;
    }
#pragma unroll
    for (int tt = 0; tt < 16; ++tt)
      bf.u[tt] = f2bf(buf[(hi * 16 + tt) * ROWSTRIDE + d0 + ln]);
    acc = __builtin_amdgcn_wmma_f32_16x16x32_bf16(false, a.v, false, bf.v,
                                                  (short)0, acc, false, false);
    __syncthreads();                            // all waves done reading chunk t
  }
  if (hi == 0) {
    const int d = d0 + ln;
#pragma unroll
    for (int r = 0; r < NHEAD; ++r)
      attnout[(((long long)b * NHEAD + r) * SEQ + i) * DHEAD + d] += acc[r];
  }
}

// ---------------- merge heads (b,h,i,d) -> bf16 (b,i,h*64+d) ----------------
__global__ void k_repack_ao(const float* __restrict__ ao, uint16_t* __restrict__ dst) {
  long long idx = (long long)blockIdx.x * blockDim.x + threadIdx.x; // NTOK*HID
  int d = (int)(idx % DHEAD);
  long long r = idx / DHEAD;
  int i = (int)(r % SEQ); r /= SEQ;
  int h = (int)(r % NHEAD);
  int b = (int)(r / NHEAD);
  dst[((long long)b * SEQ + i) * HID + h * DHEAD + d] = f2bf(ao[idx]);
}

// ---------------- fused residual add + LayerNorm ----------------
__global__ void k_add_ln(const float* __restrict__ x, const float* __restrict__ y,
                         const float* __restrict__ sc, const float* __restrict__ bi,
                         float* __restrict__ outf, uint16_t* __restrict__ outb) {
  const int lane = threadIdx.x & 31;
  const int wv = threadIdx.x >> 5;
  const long long row = (long long)blockIdx.x * (blockDim.x >> 5) + wv;
  const float* xr = x + row * HID;
  const float* yr = y + row * HID;
  float v[16]; float s = 0.0f;
#pragma unroll
  for (int t = 0; t < 16; ++t) { const int j = lane + t * 32; v[t] = xr[j] + yr[j]; s += v[t]; }
#pragma unroll
  for (int off = 16; off > 0; off >>= 1) s += __shfl_xor(s, off, 32);
  const float mean = s * (1.0f / HID);
  float q = 0.0f;
#pragma unroll
  for (int t = 0; t < 16; ++t) { const float d = v[t] - mean; q += d * d; }
#pragma unroll
  for (int off = 16; off > 0; off >>= 1) q += __shfl_xor(q, off, 32);
  const float inv = rsqrtf(q * (1.0f / HID) + LN_EPS);
#pragma unroll
  for (int t = 0; t < 16; ++t) {
    const int j = lane + t * 32;
    const float o = (v[t] - mean) * inv * sc[j] + bi[j];
    if (outf) outf[row * HID + j] = o;
    if (outb) outb[row * HID + j] = f2bf(o);
  }
}

extern "C" void kernel_launch(void* const* d_in, const int* in_sizes, int n_in,
                              void* d_out, int out_size, void* d_ws, size_t ws_size,
                              hipStream_t stream) {
  (void)in_sizes; (void)n_in; (void)out_size; (void)ws_size;
  const float* inp          = (const float*)d_in[0];
  const unsigned char* mask = (const unsigned char*)d_in[1];  // jax bool -> 1 byte
  const float* relk = (const float*)d_in[2];
  const float* relv = (const float*)d_in[3];
  const float* qkvw = (const float*)d_in[4];
  const float* qkvb = (const float*)d_in[5];
  const float* caw  = (const float*)d_in[6];
  const float* cab  = (const float*)d_in[7];
  const float* ln1s = (const float*)d_in[8];
  const float* ln1b = (const float*)d_in[9];
  const float* w1   = (const float*)d_in[10];
  const float* b1   = (const float*)d_in[11];
  const float* w2   = (const float*)d_in[12];
  const float* b2   = (const float*)d_in[13];
  const float* ln2s = (const float*)d_in[14];
  const float* ln2b = (const float*)d_in[15];
  float* out = (float*)d_out;

  char* w = (char*)d_ws;
  auto take = [&](size_t bytes) -> char* {
    char* p = w; w += (bytes + 255) & ~(size_t)255; return p;
  };
  uint16_t* Xb   = (uint16_t*)take((size_t)NTOK * HID * 2);
  uint16_t* WqbT = (uint16_t*)take((size_t)3 * HID * HID * 2);  // (3H x H)
  uint16_t* WcbT = (uint16_t*)take((size_t)HID * HID * 2);      // (H x H)
  uint16_t* W1bT = (uint16_t*)take((size_t)4 * HID * HID * 2);  // (4H x H)
  uint16_t* W2bT = (uint16_t*)take((size_t)HID * 4 * HID * 2);  // (H x 4H)
  uint16_t* QKVb = (uint16_t*)take((size_t)NTOK * 3 * HID * 2);
  uint16_t* Qb   = (uint16_t*)take((size_t)NTOK * HID * 2);
  uint16_t* Kb   = (uint16_t*)take((size_t)NTOK * HID * 2);
  uint16_t* Vtb  = (uint16_t*)take((size_t)NTOK * HID * 2);
  float*    Sc   = (float*)   take((size_t)BSZ * NHEAD * SEQ * SEQ * 4);
  uint16_t* Ab   = (uint16_t*)take((size_t)BSZ * NHEAD * SEQ * SEQ * 2);
  float*    AO   = (float*)   take((size_t)NTOK * HID * 4);
  uint16_t* AOb  = (uint16_t*)take((size_t)NTOK * HID * 2);
  float*    PR   = (float*)   take((size_t)NTOK * HID * 4);
  float*    H1   = (float*)   take((size_t)NTOK * HID * 4);
  uint16_t* H1b  = (uint16_t*)take((size_t)NTOK * HID * 2);
  uint16_t* MIDb = (uint16_t*)take((size_t)NTOK * 4 * HID * 2);
  float*    FF2  = (float*)   take((size_t)NTOK * HID * 4);

  // 1) bf16 conversions; weights transposed to N x K for contiguous B fragments
  k_cvt_bf16_v4<<<(NTOK * HID / 4) / 256, 256, 0, stream>>>(
      (const float4*)inp, (uint2*)Xb, NTOK * HID / 4);
  k_cvt_bf16_t<<<(HID * 3 * HID) / 256, 256, 0, stream>>>(qkvw, WqbT, HID, 3 * HID);
  k_cvt_bf16_t<<<(HID * HID) / 256, 256, 0, stream>>>(caw, WcbT, HID, HID);
  k_cvt_bf16_t<<<(HID * 4 * HID) / 256, 256, 0, stream>>>(w1, W1bT, HID, 4 * HID);
  k_cvt_bf16_t<<<(4 * HID * HID) / 256, 256, 0, stream>>>(w2, W2bT, 4 * HID, HID);

  // 2) QKV projection (2048x1536x512)
  { GemmP p{}; p.A = Xb; p.B = WqbT; p.Cb = QKVb; p.bias = qkvb;
    p.lda = HID; p.ldb = HID; p.ldcb = 3 * HID; p.K = HID;
    k_gemm_bf16<<<dim3(3 * HID / 64, NTOK / 32, 1), 32, 0, stream>>>(p); }

  // 3) split heads (V transposed)
  k_repack_qkv<<<(NTOK * 3 * HID) / 256, 256, 0, stream>>>(QKVb, Qb, Kb, Vtb);

  // 4) scores = Q . K^T, batched over (b,h)  (K stored j x d == N x K)
  { GemmP p{}; p.A = Qb; p.B = Kb; p.C = Sc;
    p.sA = (long long)SEQ * DHEAD; p.sB = (long long)SEQ * DHEAD; p.sC = (long long)SEQ * SEQ;
    p.lda = DHEAD; p.ldb = DHEAD; p.ldc = SEQ; p.K = DHEAD;
    k_gemm_bf16<<<dim3(SEQ / 64, SEQ / 32, BSZ * NHEAD), 32, 0, stream>>>(p); }

  // 5) scores += Q . rel_k^T (single streaming pass over 256MB)
  k_rel_scores<<<dim3(SEQ / 64, SEQ, BSZ), 32, 0, stream>>>(Qb, relk, Sc);

  // 6) mask + softmax -> bf16 attention weights
  k_softmax<<<(BSZ * NHEAD * SEQ) / 8, 256, 0, stream>>>(Sc, mask, Ab);

  // 7) out = A . V, batched over (b,h)  (V stored d x j == N x K)
  { GemmP p{}; p.A = Ab; p.B = Vtb; p.C = AO;
    p.sA = (long long)SEQ * SEQ; p.sB = (long long)DHEAD * SEQ; p.sC = (long long)SEQ * DHEAD;
    p.lda = SEQ; p.ldb = SEQ; p.ldc = DHEAD; p.K = SEQ;
    k_gemm_bf16<<<dim3(DHEAD / 64, SEQ / 32, BSZ * NHEAD), 32, 0, stream>>>(p); }

  // 8) out += A . rel_v (TDM-staged streaming pass over 256MB)
  k_rel_out<<<dim3(SEQ, BSZ, 1), 128, 0, stream>>>(Ab, relv, AO);

  // 9) merge heads -> bf16
  k_repack_ao<<<(NTOK * HID) / 256, 256, 0, stream>>>(AO, AOb);

  // 10) output projection
  { GemmP p{}; p.A = AOb; p.B = WcbT; p.C = PR; p.bias = cab;
    p.lda = HID; p.ldb = HID; p.ldc = HID; p.K = HID;
    k_gemm_bf16<<<dim3(HID / 64, NTOK / 32, 1), 32, 0, stream>>>(p); }

  // 11) h1 = LN(inputs + proj)
  k_add_ln<<<NTOK / 8, 256, 0, stream>>>(inp, PR, ln1s, ln1b, H1, H1b);

  // 12) FFN1 + ReLU (bf16 out directly)
  { GemmP p{}; p.A = H1b; p.B = W1bT; p.Cb = MIDb; p.bias = b1; p.relu = 1;
    p.lda = HID; p.ldb = HID; p.ldcb = 4 * HID; p.K = HID;
    k_gemm_bf16<<<dim3(4 * HID / 64, NTOK / 32, 1), 32, 0, stream>>>(p); }

  // 13) FFN2
  { GemmP p{}; p.A = MIDb; p.B = W2bT; p.C = FF2; p.bias = b2;
    p.lda = 4 * HID; p.ldb = 4 * HID; p.ldc = HID; p.K = 4 * HID;
    k_gemm_bf16<<<dim3(HID / 64, NTOK / 32, 1), 32, 0, stream>>>(p); }

  // 14) out = LN(h1 + ff)
  k_add_ln<<<NTOK / 8, 256, 0, stream>>>(H1, FF2, ln2s, ln2b, out, nullptr);
}
